// SelfAttention_74302934221032
// MI455X (gfx1250) — matrix operand order repaired and checked by reading, not compiled, not run on hardware
//
#include <hip/hip_runtime.h>

// ---------------------------------------------------------------------------
// Self-attention for MI455X (gfx1250, wave32, WMMA).
//   x:[4096,1024] f32, mask:[4096,4096] i32, Wq/Wk/Wv:[1024,1024] f32
//   out:[4096,1024] f32
// bf16 WMMA (f32 accumulate) for all GEMMs; fused attention keeps the
// 16x4096 score row-block in CDNA5's 320KB LDS. Register-blocked tiles
// raise per-wave arithmetic intensity; LDS strides padded to kill bank
// conflicts on the WMMA A-operand reads.
// ---------------------------------------------------------------------------

typedef __attribute__((ext_vector_type(16))) __bf16 v16bf;
typedef __attribute__((ext_vector_type(8)))  float  v8f;

#define N_TOK 4096
#define EM    1024
#define SCALE 0.03125f      // 1/sqrt(1024)

// padded LDS strides (dwords mod 64 == 4 -> lane l starts at bank 4l)
#define PSTR  4100          // P row stride in floats   (4096 + 4)
#define QSTR  1032          // qs row stride in bf16    (1024 + 8)

// ---- LDS layout for attention kernel (dynamic LDS) ------------------------
#define LDS_P    0                       // float P[16][4100]  = 262400 B
#define LDS_QS   262400                  // bf16 qs[16][1032]  =  33024 B
#define LDS_RED  295424                  // float red[256]     =   1024 B
#define LDS_RSI  296448                  // float rsi[16]      =     64 B
#define LDS_SZ   296512

// ---------------------------------------------------------------------------
__global__ void cast_f32_bf16(const float* __restrict__ src,
                              __bf16* __restrict__ dst, int n) {
  int i = blockIdx.x * blockDim.x + threadIdx.x;
  if (i < (n >> 1)) {
    float2 f = ((const float2*)src)[i];
    __bf16 b0 = (__bf16)f.x;
    __bf16 b1 = (__bf16)f.y;
    ushort2 u;
    u.x = __builtin_bit_cast(unsigned short, b0);
    u.y = __builtin_bit_cast(unsigned short, b1);
    ((ushort2*)dst)[i] = u;
  }
}

// ---------------------------------------------------------------------------
// q = x@Wq^T, k = x@Wk^T, v = x@Wv^T (v stored transposed as vt[e][i]).
// Register blocking: each wave owns a 32x64 output block = 2x4 WMMA tiles.
// Per K-step: 2 A-tile + 4 B-tile loads feed 8 WMMAs (24 B/lane/WMMA).
// blockDim=256 (8 waves); grid = (256, 3).
// ---------------------------------------------------------------------------
__launch_bounds__(256, 2)
__global__ void qkv_gemm(const __bf16* __restrict__ xb,
                         const __bf16* __restrict__ wqb,
                         const __bf16* __restrict__ wkb,
                         const __bf16* __restrict__ wvb,
                         __bf16* __restrict__ qb,
                         __bf16* __restrict__ kb,
                         __bf16* __restrict__ vt) {
  const int lane  = threadIdx.x & 31;
  const int wave  = threadIdx.x >> 5;
  const int z     = blockIdx.y;                 // 0=q 1=k 2=v
  const int job   = blockIdx.x * 8 + wave;      // 0..2047
  const int tr    = job >> 4;                   // 32-row strip  (128)
  const int tc    = job & 15;                   // 64-col strip  (16)
  const int lhalf = lane >> 4;
  const int lmod  = lane & 15;
  const int r0    = tr * 32;
  const int c0    = tc * 64;

  const __bf16* W = (z == 0) ? wqb : (z == 1) ? wkb : wvb;

  const __bf16* ap0 = xb + (r0 + 0  + lmod) * EM + 16 * lhalf;
  const __bf16* ap1 = xb + (r0 + 16 + lmod) * EM + 16 * lhalf;
  const __bf16* bp0 = W + (c0 + 0  + lmod) * EM + 16 * lhalf;
  const __bf16* bp1 = W + (c0 + 16 + lmod) * EM + 16 * lhalf;
  const __bf16* bp2 = W + (c0 + 32 + lmod) * EM + 16 * lhalf;
  const __bf16* bp3 = W + (c0 + 48 + lmod) * EM + 16 * lhalf;

  const v8f zero = {0.f, 0.f, 0.f, 0.f, 0.f, 0.f, 0.f, 0.f};
  v8f acc[2][4];
  for (int i = 0; i < 2; ++i)
    for (int j = 0; j < 4; ++j) acc[i][j] = zero;

  for (int ec = 0; ec < EM / 32; ++ec) {
    union { uint4 q[2]; v16bf v; } A0, A1;
    A0.q[0] = *(const uint4*)(ap0);
    A0.q[1] = *(const uint4*)(ap0 + 8);
    A1.q[0] = *(const uint4*)(ap1);
    A1.q[1] = *(const uint4*)(ap1 + 8);
    v16bf B0 = *(const v16bf*)(bp0);
    v16bf B1 = *(const v16bf*)(bp1);
    v16bf B2 = *(const v16bf*)(bp2);
    v16bf B3 = *(const v16bf*)(bp3);
    acc[0][0] = __builtin_amdgcn_wmma_f32_16x16x32_bf16(false, A0.v, false, B0, (short)0, acc[0][0], false, false);
    acc[0][1] = __builtin_amdgcn_wmma_f32_16x16x32_bf16(false, A0.v, false, B1, (short)0, acc[0][1], false, false);
    acc[0][2] = __builtin_amdgcn_wmma_f32_16x16x32_bf16(false, A0.v, false, B2, (short)0, acc[0][2], false, false);
    acc[0][3] = __builtin_amdgcn_wmma_f32_16x16x32_bf16(false, A0.v, false, B3, (short)0, acc[0][3], false, false);
    acc[1][0] = __builtin_amdgcn_wmma_f32_16x16x32_bf16(false, A1.v, false, B0, (short)0, acc[1][0], false, false);
    acc[1][1] = __builtin_amdgcn_wmma_f32_16x16x32_bf16(false, A1.v, false, B1, (short)0, acc[1][1], false, false);
    acc[1][2] = __builtin_amdgcn_wmma_f32_16x16x32_bf16(false, A1.v, false, B2, (short)0, acc[1][2], false, false);
    acc[1][3] = __builtin_amdgcn_wmma_f32_16x16x32_bf16(false, A1.v, false, B3, (short)0, acc[1][3], false, false);
    ap0 += 32; ap1 += 32; bp0 += 32; bp1 += 32; bp2 += 32; bp3 += 32;
  }

  if (z < 2) {
    __bf16* out = (z == 0) ? qb : kb;
    for (int rt = 0; rt < 2; ++rt) {
      const int rbase = r0 + rt * 16 + 8 * lhalf;
      for (int ct = 0; ct < 4; ++ct) {
        const int col = c0 + ct * 16 + lmod;
        for (int r = 0; r < 8; ++r)
          out[(rbase + r) * EM + col] = (__bf16)acc[rt][ct][r];
      }
    }
  } else {
    // transposed store: vt[e][i], 8 consecutive i per lane -> one b128 store
    for (int rt = 0; rt < 2; ++rt) {
      const int ibase = r0 + rt * 16 + 8 * lhalf;
      for (int ct = 0; ct < 4; ++ct) {
        const int e = c0 + ct * 16 + lmod;
        union { uint4 q; __bf16 h[8]; } o;
        for (int r = 0; r < 8; ++r) o.h[r] = (__bf16)acc[rt][ct][r];
        *(uint4*)(vt + e * N_TOK + ibase) = o.q;
      }
    }
  }
}

// ---------------------------------------------------------------------------
// Fused attention: one 16-row query block per workgroup (256 thr = 8 waves).
//   stage1: S = q@k^T (masked, scaled) -> LDS; wave owns 512 key cols,
//           processed 4 tiles at a time so A (LDS) is reused x4.
//   stage2: softmax (unnormalized exp in place; 16-strided col partition)
//   stage3: out = P@v via vt; wave owns 128 E cols, A reused x8.
// ---------------------------------------------------------------------------
__launch_bounds__(256, 1)
__global__ void attn_kernel(const __bf16* __restrict__ qb,
                            const __bf16* __restrict__ kb,
                            const __bf16* __restrict__ vt,
                            const int*    __restrict__ mask,
                            float*        __restrict__ out) {
  extern __shared__ char lds[];
  float*  P   = (float*)(lds + LDS_P);
  __bf16* qs  = (__bf16*)(lds + LDS_QS);
  float*  red = (float*)(lds + LDS_RED);
  float*  rsi = (float*)(lds + LDS_RSI);

  const int tid   = threadIdx.x;
  const int lane  = tid & 31;
  const int wave  = tid >> 5;
  const int lhalf = lane >> 4;
  const int lmod  = lane & 15;
  const int row0  = blockIdx.x * 16;

  // ---- stage the q block into LDS (padded row stride) ----
  {
    const uint* src = (const uint*)(qb + row0 * EM);
    uint* dst = (uint*)qs;
    for (int i = tid; i < 16 * (EM / 2); i += 256) {
      const int r = i >> 9;             // row
      const int c = i & 511;            // dword in row
      dst[r * (QSTR / 2) + c] = src[i];
    }
  }
  __syncthreads();

  const v8f zero = {0.f, 0.f, 0.f, 0.f, 0.f, 0.f, 0.f, 0.f};

  // ---- stage 1: scores, 4 key tiles per pass ----
  for (int tg = 0; tg < 8; ++tg) {
    const int col0 = wave * 512 + tg * 64;
    v8f acc[4];
    for (int j = 0; j < 4; ++j) acc[j] = zero;

    const __bf16* ap  = qs + lmod * QSTR + 16 * lhalf;                // LDS
    const __bf16* bp0 = kb + (col0 + 0  + lmod) * EM + 16 * lhalf;    // global
    const __bf16* bp1 = kb + (col0 + 16 + lmod) * EM + 16 * lhalf;
    const __bf16* bp2 = kb + (col0 + 32 + lmod) * EM + 16 * lhalf;
    const __bf16* bp3 = kb + (col0 + 48 + lmod) * EM + 16 * lhalf;

    for (int ec = 0; ec < EM / 32; ++ec) {
      union { uint4 q[2]; v16bf v; } A;
      A.q[0] = *(const uint4*)(ap);
      A.q[1] = *(const uint4*)(ap + 8);
      v16bf B0 = *(const v16bf*)(bp0);
      v16bf B1 = *(const v16bf*)(bp1);
      v16bf B2 = *(const v16bf*)(bp2);
      v16bf B3 = *(const v16bf*)(bp3);
      acc[0] = __builtin_amdgcn_wmma_f32_16x16x32_bf16(false, A.v, false, B0, (short)0, acc[0], false, false);
      acc[1] = __builtin_amdgcn_wmma_f32_16x16x32_bf16(false, A.v, false, B1, (short)0, acc[1], false, false);
      acc[2] = __builtin_amdgcn_wmma_f32_16x16x32_bf16(false, A.v, false, B2, (short)0, acc[2], false, false);
      acc[3] = __builtin_amdgcn_wmma_f32_16x16x32_bf16(false, A.v, false, B3, (short)0, acc[3], false, false);
      ap += 32; bp0 += 32; bp1 += 32; bp2 += 32; bp3 += 32;
    }

    const int mloc = 8 * lhalf;
    for (int ct = 0; ct < 4; ++ct) {
      const int colg = col0 + ct * 16 + lmod;
      for (int r = 0; r < 8; ++r) {
        const int m = mask[(row0 + mloc + r) * N_TOK + colg];
        float v = acc[ct][r];
        v = (m == 0 ? -1e20f : v) * SCALE;     // mask BEFORE scale (ref order)
        P[(mloc + r) * PSTR + colg] = v;
      }
    }
  }
  __syncthreads();

  // ---- stage 2: softmax (16 threads/row; 16-strided cols for bank spread) --
  {
    const int row = tid >> 4;
    const int sub = tid & 15;
    float* p = P + row * PSTR;
    float m = -3.4e38f;
    for (int i = 0; i < 256; ++i) m = fmaxf(m, p[sub + 16 * i]);
    red[tid] = m;
    __syncthreads();
    float rm = -3.4e38f;
    for (int k = 0; k < 16; ++k) rm = fmaxf(rm, red[(row << 4) + k]);
    float s = 0.f;
    for (int i = 0; i < 256; ++i) {
      const int c = sub + 16 * i;
      float e = __expf(p[c] - rm);
      p[c] = e;                               // unnormalized prob, in place
      s += e;
    }
    __syncthreads();
    red[tid] = s;
    __syncthreads();
    if (sub == 0) {
      float tot = 0.f;
      for (int k = 0; k < 16; ++k) tot += red[(row << 4) + k];
      rsi[row] = 1.0f / tot;
    }
  }
  __syncthreads();

  // ---- stage 3: out = P @ v  (wave owns E columns [wave*128, +128)) ----
  {
    v8f acc[8];
    for (int t = 0; t < 8; ++t) acc[t] = zero;
    const int e0 = wave * 128;

    for (int jc = 0; jc < N_TOK / 32; ++jc) {
      const int j0 = jc * 32;
      // A tile from LDS P, converted f32 -> bf16 (reused by 8 WMMAs)
      const float* pr = P + lmod * PSTR + j0 + 16 * lhalf;
      float4 f0 = *(const float4*)(pr);
      float4 f1 = *(const float4*)(pr + 4);
      float4 f2 = *(const float4*)(pr + 8);
      float4 f3 = *(const float4*)(pr + 12);
      v16bf A;
      A[0]=(__bf16)f0.x;  A[1]=(__bf16)f0.y;  A[2]=(__bf16)f0.z;  A[3]=(__bf16)f0.w;
      A[4]=(__bf16)f1.x;  A[5]=(__bf16)f1.y;  A[6]=(__bf16)f1.z;  A[7]=(__bf16)f1.w;
      A[8]=(__bf16)f2.x;  A[9]=(__bf16)f2.y;  A[10]=(__bf16)f2.z; A[11]=(__bf16)f2.w;
      A[12]=(__bf16)f3.x; A[13]=(__bf16)f3.y; A[14]=(__bf16)f3.z; A[15]=(__bf16)f3.w;
      for (int t = 0; t < 8; ++t) {
        const __bf16* bp = vt + (e0 + t * 16 + lmod) * N_TOK + j0 + 16 * lhalf;
        v16bf B = *(const v16bf*)(bp);
        acc[t] = __builtin_amdgcn_wmma_f32_16x16x32_bf16(false, A, false, B,
                                                         (short)0, acc[t],
                                                         false, false);
      }
    }

    const int mloc = 8 * lhalf;
    float rs[8];
    for (int r = 0; r < 8; ++r) rs[r] = rsi[mloc + r];
    for (int t = 0; t < 8; ++t) {
      const int col = e0 + t * 16 + lmod;
      for (int r = 0; r < 8; ++r)
        out[(row0 + mloc + r) * EM + col] = acc[t][r] * rs[r];
    }
  }
}

// ---------------------------------------------------------------------------
extern "C" void kernel_launch(void* const* d_in, const int* in_sizes, int n_in,
                              void* d_out, int out_size, void* d_ws, size_t ws_size,
                              hipStream_t stream) {
  const float* x    = (const float*)d_in[0];
  const int*   mask = (const int*)  d_in[1];
  const float* Wq   = (const float*)d_in[2];
  const float* Wk   = (const float*)d_in[3];
  const float* Wv   = (const float*)d_in[4];
  float* out = (float*)d_out;

  // workspace layout (bytes)
  char* ws = (char*)d_ws;
  __bf16* xb  = (__bf16*)(ws + 0);          // 8 MiB
  __bf16* wqb = (__bf16*)(ws + 8388608);    // 2 MiB
  __bf16* wkb = (__bf16*)(ws + 10485760);   // 2 MiB
  __bf16* wvb = (__bf16*)(ws + 12582912);   // 2 MiB
  __bf16* qb  = (__bf16*)(ws + 14680064);   // 8 MiB
  __bf16* kb  = (__bf16*)(ws + 23068672);   // 8 MiB
  __bf16* vt  = (__bf16*)(ws + 31457280);   // 8 MiB  (total ~38 MiB)

  const int NX = N_TOK * EM;   // 4194304
  const int NW = EM * EM;      // 1048576
  cast_f32_bf16<<<(NX / 2 + 255) / 256, 256, 0, stream>>>(x,  xb,  NX);
  cast_f32_bf16<<<(NW / 2 + 255) / 256, 256, 0, stream>>>(Wq, wqb, NW);
  cast_f32_bf16<<<(NW / 2 + 255) / 256, 256, 0, stream>>>(Wk, wkb, NW);
  cast_f32_bf16<<<(NW / 2 + 255) / 256, 256, 0, stream>>>(Wv, wvb, NW);

  qkv_gemm<<<dim3(256, 3), 256, 0, stream>>>(xb, wqb, wkb, wvb, qb, kb, vt);

  // 296512 B dynamic LDS (CDNA5 WGP has 320KB); raise the per-kernel cap.
  hipFuncSetAttribute((const void*)attn_kernel,
                      hipFuncAttributeMaxDynamicSharedMemorySize, LDS_SZ);
  attn_kernel<<<N_TOK / 16, 256, LDS_SZ, stream>>>(qb, kb, vt, mask, out);
}
